// MultiScaleGeoEncoder_13134009991335
// MI455X (gfx1250) — compile-verified, matrix operand order checked by reading, not run on hardware
//
#include <hip/hip_runtime.h>

typedef float v8f __attribute__((ext_vector_type(8)));
typedef __bf16 v16bf __attribute__((ext_vector_type(16)));

#define B_   8
#define N_   2048
#define FIN_ 128
#define H_   128
#define S_   2
#define BN_  (B_ * N_)

union Frag16 { unsigned u[8]; v16bf v; };

__device__ __forceinline__ unsigned short f2bf(float x) {
  union { float f; unsigned u; } c; c.f = x;
  unsigned r = c.u + 0x7fffu + ((c.u >> 16) & 1u);   // round-to-nearest-even
  return (unsigned short)(r >> 16);
}

// K base (even half) for A/B bf16 16x16x32 operand: VGPR-half v (0..7), lane group g (0..1)
__device__ __forceinline__ int kbase(int v, int g) {
  return ((v >> 2) << 4) + (g << 3) + ((v & 3) << 1);
}

// Async global -> LDS copies (CDNA5). LDS operand = wave-relative LDS byte
// address (low 32 bits of the generic pointer, per ISA aperture rules).
__device__ __forceinline__ void async_b128(void* lds, const void* gptr) {
  unsigned l = (unsigned)(size_t)lds;
  asm volatile("global_load_async_to_lds_b128 %0, %1, off"
               :: "v"(l), "v"(gptr) : "memory");
}
__device__ __forceinline__ void async_b32(void* lds, const void* gptr) {
  unsigned l = (unsigned)(size_t)lds;
  asm volatile("global_load_async_to_lds_b32 %0, %1, off"
               :: "v"(l), "v"(gptr) : "memory");
}

// ---------------------------------------------------------------------------
// Stage A: h[s] = X @ W[s]^T + b[s], stored TRANSPOSED per (s,b): [col][n]
// (so stage B can async-copy B-operand tiles without a transpose), plus
// si = h@a_i, sj = h@a_j.  grid (BN/128, S), 256 threads.
// ---------------------------------------------------------------------------
__global__ __launch_bounds__(256) void geo_h_kernel(
    const float* __restrict__ X, const float* __restrict__ W_w,
    const float* __restrict__ W_b, const float* __restrict__ a_i,
    const float* __restrict__ a_j, unsigned short* __restrict__ h_t,
    float* __restrict__ si_ws, float* __restrict__ sj_ws)
{
  __shared__ __align__(16) unsigned short X_lds[128][136];  // [row][k] bf16
  __shared__ __align__(16) unsigned short W_lds[128][136];  // [out col][k] bf16
  const int t = threadIdx.x;
  const int w = t >> 5, l = t & 31, g = l >> 4, ln = l & 15;
  const int s = blockIdx.y;
  const size_t m0 = (size_t)blockIdx.x * 128;
  const int b = (int)(m0 / N_);
  const int n0 = (int)(m0 % N_);

  #pragma unroll
  for (int i = 0; i < 64; ++i) {
    const int gi = t + i * 256;                        // 16384 elements each
    W_lds[gi >> 7][gi & 127] = f2bf(W_w[(size_t)s * (H_ * FIN_) + gi]);
    X_lds[gi >> 7][gi & 127] = f2bf(X[m0 * FIN_ + gi]);
  }
  __syncthreads();

  v8f acc[8] = {};
  const int row = w * 16 + ln;
  #pragma unroll
  for (int kk = 0; kk < 4; ++kk) {
    Frag16 Af;
    #pragma unroll
    for (int v = 0; v < 8; ++v)
      Af.u[v] = *(const unsigned*)&X_lds[row][kk * 32 + kbase(v, g)];
    #pragma unroll
    for (int tt = 0; tt < 8; ++tt) {
      Frag16 Bf;
      #pragma unroll
      for (int v = 0; v < 8; ++v)
        Bf.u[v] = *(const unsigned*)&W_lds[tt * 16 + ln][kk * 32 + kbase(v, g)];
      acc[tt] = __builtin_amdgcn_wmma_f32_16x16x32_bf16(
          false, Af.v, false, Bf.v, (short)0, acc[tt], false, false);
    }
  }

  // bias, transposed packed h store (8 consecutive n per lane -> one b128),
  // and per-row dots with a_i / a_j
  float pi[8] = {}, pj[8] = {};
  const int nbase = n0 + w * 16 + 8 * g;   // rows r=0..7 are n = nbase..nbase+7
  #pragma unroll
  for (int tt = 0; tt < 8; ++tt) {
    const int col = tt * 16 + ln;
    const float bias = W_b[s * H_ + col];
    const float aiv = a_i[s * H_ + col];
    const float ajv = a_j[s * H_ + col];
    float hv[8];
    #pragma unroll
    for (int r = 0; r < 8; ++r) {
      hv[r] = acc[tt][r] + bias;
      pi[r] += hv[r] * aiv;
      pj[r] += hv[r] * ajv;
    }
    uint4 pk;
    pk.x = (unsigned)f2bf(hv[0]) | ((unsigned)f2bf(hv[1]) << 16);
    pk.y = (unsigned)f2bf(hv[2]) | ((unsigned)f2bf(hv[3]) << 16);
    pk.z = (unsigned)f2bf(hv[4]) | ((unsigned)f2bf(hv[5]) << 16);
    pk.w = (unsigned)f2bf(hv[6]) | ((unsigned)f2bf(hv[7]) << 16);
    *(uint4*)&h_t[((size_t)(s * B_ + b) * H_ + col) * N_ + nbase] = pk;
  }
  #pragma unroll
  for (int r = 0; r < 8; ++r) {
    #pragma unroll
    for (int msk = 1; msk < 16; msk <<= 1) {
      pi[r] += __shfl_xor(pi[r], msk, 32);
      pj[r] += __shfl_xor(pj[r], msk, 32);
    }
  }
  if (ln == 0) {
    #pragma unroll
    for (int r = 0; r < 8; ++r) {
      const size_t mrow = m0 + (size_t)(w * 16 + r + 8 * g);
      si_ws[(size_t)s * BN_ + mrow] = pi[r];
      sj_ws[(size_t)s * BN_ + mrow] = pj[r];
    }
  }
}

// ---------------------------------------------------------------------------
// Stage B: fused gated attention, online softmax, async double-buffered
// global->LDS staging (7 async instructions per wave per tile).
// grid (N/128, B, S), 256 threads; wave w owns 16 rows, j tiles of 32.
// ---------------------------------------------------------------------------
__global__ __launch_bounds__(256) void geo_attn_kernel(
    const unsigned short* __restrict__ h_t, const float* __restrict__ si_ws,
    const float* __restrict__ sj_ws, const float* __restrict__ dist,
    const float* __restrict__ sigmas, const float* __restrict__ a_b,
    const float* __restrict__ thr, unsigned short* __restrict__ hcat)
{
  __shared__ __align__(16) unsigned short h_lds[2][128][40]; // [buf][H col][j]
  __shared__ __align__(16) float d_lds[2][128][36];          // [buf][i][j]
  __shared__ __align__(16) float sj_lds[2][8][32];           // [buf][wave][j]
  const int t = threadIdx.x;
  const int w = t >> 5, l = t & 31, g = l >> 4, ln = l & 15;
  const int s = blockIdx.z, b = blockIdx.y;
  const int i0 = blockIdx.x * 128;
  const float sig = sigmas[s];
  const float inv2s2 = 1.0f / (2.0f * sig * sig);
  const float th = thr[s], ab = a_b[s];
  const int i_local = w * 16 + ln;
  const int i_glob = i0 + i_local;
  const float si_m = si_ws[(size_t)s * BN_ + (size_t)b * N_ + i_glob];
  const size_t hrow = (size_t)(s * B_ + b) * H_;           // h_t row base (cols)
  const float* sjb = sj_ws + (size_t)s * BN_ + (size_t)b * N_;
  float m_run = -1e30f, s_run = 0.0f;
  v8f acc[8] = {};

  // issue the 7 per-wave async copies for tile starting at j0 into buffer buf
  auto issue_tile = [&](int j0, int buf) {
    #pragma unroll
    for (int i = 0; i < 2; ++i) {            // h tile: 128 rows x 64B
      const int ch = t + i * 256;            // 512 b128 chunks
      const int c = ch >> 2, q = ch & 3;
      async_b128(&h_lds[buf][c][q * 8],
                 (const char*)h_t + ((hrow + c) * N_ + j0) * 2 + q * 16);
    }
    #pragma unroll
    for (int i = 0; i < 4; ++i) {            // dist tile: 128 rows x 128B
      const int ch = t + i * 256;            // 1024 b128 chunks
      const int r = ch >> 3, q = ch & 7;
      async_b128(&d_lds[buf][r][q * 4],
                 dist + (size_t)(i0 + r) * N_ + j0 + q * 4);
    }
    async_b32(&sj_lds[buf][w][l & 31], sjb + j0 + l);  // per-wave sj slice
  };

  issue_tile(0, 0);
  for (int jt = 0; jt < N_ / 32; ++jt) {
    const int j0 = jt * 32, buf = jt & 1;
    if (jt + 1 < N_ / 32) {
      issue_tile((jt + 1) * 32, buf ^ 1);
      asm volatile("s_wait_asynccnt 0x7" ::: "memory");  // tile jt retired
    } else {
      asm volatile("s_wait_asynccnt 0x0" ::: "memory");
    }
    __syncthreads();

    float pv[16];
    float tmax = -1e30f;
    #pragma unroll
    for (int idx = 0; idx < 16; ++idx) {  // logits in A-operand slot order
      const int jl = kbase(idx >> 1, g) + (idx & 1);
      const float d = d_lds[buf][i_local][jl];
      const float dd = (i_glob == (j0 + jl)) ? 1.0f : d;  // diag=1 for 1/d bias
      float ev = si_m + sj_lds[buf][w][jl] + ab;
      ev = (ev >= 0.0f) ? ev : 0.1f * ev;                 // LeakyReLU
      ev += 1.0f / (dd + 1e-5f);
      const float A = __expf(-d * d * inv2s2);            // Gaussian kernel
      const float mask = 1.0f / (1.0f + __expf(-10.0f * (A - th)));
      ev *= mask;
      pv[idx] = ev;
      tmax = fmaxf(tmax, ev);
    }
    tmax = fmaxf(tmax, __shfl_xor(tmax, 16, 32));  // both K-halves of a row
    const float m_new = fmaxf(m_run, tmax);
    const float alpha = __expf(m_run - m_new);
    float rsum = 0.0f;
    Frag16 P;
    #pragma unroll
    for (int v = 0; v < 8; ++v) {
      const float p0 = __expf(pv[2 * v] - m_new);
      const float p1 = __expf(pv[2 * v + 1] - m_new);
      rsum += p0 + p1;
      P.u[v] = (unsigned)f2bf(p0) | ((unsigned)f2bf(p1) << 16);
    }
    rsum += __shfl_xor(rsum, 16, 32);
    s_run = s_run * alpha + rsum;
    m_run = m_new;

    #pragma unroll
    for (int r = 0; r < 8; ++r) {   // rescale: factor for row r+8g from lane r+8g
      const float ar = __shfl(alpha, r + 8 * g, 32);
      #pragma unroll
      for (int tt = 0; tt < 8; ++tt) acc[tt][r] *= ar;
    }
    #pragma unroll
    for (int tt = 0; tt < 8; ++tt) {
      Frag16 Bf;
      #pragma unroll
      for (int v = 0; v < 8; ++v)
        Bf.u[v] = *(const unsigned*)&h_lds[buf][tt * 16 + ln][kbase(v, g)];
      acc[tt] = __builtin_amdgcn_wmma_f32_16x16x32_bf16(
          false, P.v, false, Bf.v, (short)0, acc[tt], false, false);
    }
    __syncthreads();   // all waves done reading buf before it is re-filled
  }

  const float inv = 1.0f / s_run;
  #pragma unroll
  for (int r = 0; r < 8; ++r) {
    const float ir = __shfl(inv, r + 8 * g, 32);
    const size_t row = (size_t)b * N_ + i0 + w * 16 + r + 8 * g;
    #pragma unroll
    for (int tt = 0; tt < 8; ++tt)
      hcat[row * (S_ * H_) + s * H_ + tt * 16 + ln] = f2bf(acc[tt][r] * ir);
  }
}

// ---------------------------------------------------------------------------
// Stage C: out = hcat(bf16) @ fus_w^T + fus_b   (16384 x 256 -> 128)
// grid (BN/128), 256 threads.
// ---------------------------------------------------------------------------
__global__ __launch_bounds__(256) void geo_fuse_kernel(
    const unsigned short* __restrict__ hcat, const float* __restrict__ fus_w,
    const float* __restrict__ fus_b, float* __restrict__ out)
{
  __shared__ __align__(16) unsigned short a_lds[128][264];  // [row][k]
  __shared__ __align__(16) unsigned short w_lds[128][264];  // [out col][k]
  const int t = threadIdx.x;
  const int w = t >> 5, l = t & 31, g = l >> 4, ln = l & 15;
  const size_t m0 = (size_t)blockIdx.x * 128;

  #pragma unroll
  for (int i = 0; i < 128; ++i) {             // 128x256 f32 -> bf16
    const int gi = t + i * 256;
    w_lds[gi >> 8][gi & 255] = f2bf(fus_w[gi]);
  }
  const unsigned* src = (const unsigned*)(hcat + m0 * (S_ * H_));
  #pragma unroll
  for (int i = 0; i < 64; ++i) {              // 16384 bf16-pairs
    const int gi = t + i * 256;
    *(unsigned*)&a_lds[gi >> 7][(gi & 127) * 2] = src[gi];
  }
  __syncthreads();

  v8f acc[8] = {};
  #pragma unroll
  for (int kk = 0; kk < 8; ++kk) {
    Frag16 Af;
    #pragma unroll
    for (int v = 0; v < 8; ++v)
      Af.u[v] = *(const unsigned*)&a_lds[w * 16 + ln][kk * 32 + kbase(v, g)];
    #pragma unroll
    for (int tt = 0; tt < 8; ++tt) {
      Frag16 Bf;
      #pragma unroll
      for (int v = 0; v < 8; ++v)
        Bf.u[v] = *(const unsigned*)&w_lds[tt * 16 + ln][kk * 32 + kbase(v, g)];
      acc[tt] = __builtin_amdgcn_wmma_f32_16x16x32_bf16(
          false, Af.v, false, Bf.v, (short)0, acc[tt], false, false);
    }
  }
  #pragma unroll
  for (int tt = 0; tt < 8; ++tt) {
    const int col = tt * 16 + ln;
    const float bias = fus_b[col];
    #pragma unroll
    for (int r = 0; r < 8; ++r) {
      const size_t row = m0 + (size_t)(w * 16 + r + 8 * g);
      out[row * H_ + col] = acc[tt][r] + bias;
    }
  }
}

// ---------------------------------------------------------------------------
extern "C" void kernel_launch(void* const* d_in, const int* in_sizes, int n_in,
                              void* d_out, int out_size, void* d_ws, size_t ws_size,
                              hipStream_t stream) {
  (void)in_sizes; (void)n_in; (void)out_size; (void)ws_size;
  const float* X      = (const float*)d_in[0];
  const float* dist   = (const float*)d_in[1];
  const float* sigmas = (const float*)d_in[2];
  const float* W_w    = (const float*)d_in[3];
  const float* W_b    = (const float*)d_in[4];
  const float* a_i    = (const float*)d_in[5];
  const float* a_j    = (const float*)d_in[6];
  const float* a_b    = (const float*)d_in[7];
  const float* thr    = (const float*)d_in[8];
  const float* fus_w  = (const float*)d_in[9];
  const float* fus_b  = (const float*)d_in[10];
  float* out = (float*)d_out;

  char* ws = (char*)d_ws;
  // workspace layout (bytes):
  //   h_t : S*BN*H bf16 (transposed per (s,b): [col][n]) = 8,388,608
  //   si  : S*BN f32 =   131,072
  //   sj  : S*BN f32 =   131,072
  //   hcat: BN*S*H bf16 = 8,388,608
  unsigned short* h_t   = (unsigned short*)(ws);
  float*          si_ws = (float*)(ws + 8388608);
  float*          sj_ws = (float*)(ws + 8388608 + 131072);
  unsigned short* hcat  = (unsigned short*)(ws + 8388608 + 262144);

  geo_h_kernel<<<dim3(BN_ / 128, S_), dim3(256), 0, stream>>>(
      X, W_w, W_b, a_i, a_j, h_t, si_ws, sj_ws);
  geo_attn_kernel<<<dim3(N_ / 128, B_, S_), dim3(256), 0, stream>>>(
      h_t, si_ws, sj_ws, dist, sigmas, a_b, thr, hcat);
  geo_fuse_kernel<<<dim3(BN_ / 128), dim3(256), 0, stream>>>(
      hcat, fus_w, fus_b, out);
}
